// Aggre_social_27814208209714
// MI455X (gfx1250) — compile-verified, matrix operand order checked by reading, not run on hardware
//
#include <hip/hip_runtime.h>
#include <hip/hip_bf16.h>

// Problem constants (match reference)
#define NUU 50000
#define NII 50000
#define NRR 5
#define DD  64
#define BB  512
#define LSS 32
#define LII 64
#define SPB 4                 // slots per block in kernel 1
#define NEGV (-1e9f)

typedef __attribute__((ext_vector_type(16))) __bf16 v16bf;
typedef __attribute__((ext_vector_type(8)))  float  v8f;

union Frag16 { v16bf v; unsigned short u[16]; };

// f32 -> bf16: let the compiler select the native conversion (RNE).
__device__ __forceinline__ unsigned short f2b(float f) {
    __bf16 b = (__bf16)f;
    return __builtin_bit_cast(unsigned short, b);
}
__device__ __forceinline__ float b2f(unsigned short s) {
    unsigned u = ((unsigned)s) << 16;
    return __builtin_bit_cast(float, u);
}

// D[M x 64] = relu(X[M x K] @ W[K x 64] + bias[64])
// X: LDS bf16 (ushort) row-major ld=ldx; W: LDS bf16 row-major ld=64;
// Dst: LDS bf16 row-major ld=ldd.  M in {32,64}, K in {64,128}.
// v_wmma_f32_16x16x32_bf16, wave32 layouts:
//   A 16x32: lanes 0-15 rows M; per-vgpr K pair kb = (j>=4)*16 + (j&3)*2 + hi*8
//   B 32x16: lane lo = column, same K striping
//   C 16x16: vgpr r -> row r (lanes 0-15) / r+8 (lanes 16-31), col = lane&15
// Two independent accumulators (even/odd k-steps) break the WMMA->WMMA RAW
// chain so the scheduler can interleave instead of inserting v_nops.
// Each wave keeps a fixed n0 (= wave*16) so B fragments hoist across m-tiles.
__device__ __forceinline__ void gemm_relu(const unsigned short* X, int ldx,
                                          int M, int K,
                                          const unsigned short* W,
                                          const float* bias,
                                          unsigned short* Dst, int ldd,
                                          int tid) {
    const int wave = tid >> 5;
    const int lane = tid & 31;
    const int lo = lane & 15;
    const int hi = lane >> 4;
    const int ntiles = (M >> 4) << 2;
    for (int t = wave; t < ntiles; t += 4) {
        const int m0 = (t >> 2) << 4;
        const int n0 = (t & 3) << 4;
        v8f acc0 = {}, acc1 = {};
        for (int k0 = 0; k0 < K; k0 += 64) {
            Frag16 a0, b0, a1, b1;
#pragma unroll
            for (int j = 0; j < 8; ++j) {
                const int kb = ((j >> 2) << 4) | ((j & 3) << 1) | (hi << 3);
                const unsigned short* xr0 = X + (m0 + lo) * ldx + k0 + kb;
                a0.u[2 * j]     = xr0[0];
                a0.u[2 * j + 1] = xr0[1];
                a1.u[2 * j]     = xr0[32];
                a1.u[2 * j + 1] = xr0[33];
                const unsigned short* wr0 = W + (k0 + kb) * 64 + n0 + lo;
                b0.u[2 * j]     = wr0[0];
                b0.u[2 * j + 1] = wr0[64];
                b1.u[2 * j]     = wr0[32 * 64];
                b1.u[2 * j + 1] = wr0[33 * 64];
            }
            acc0 = __builtin_amdgcn_wmma_f32_16x16x32_bf16(
                false, a0.v, false, b0.v, (short)0, acc0, false, false);
            acc1 = __builtin_amdgcn_wmma_f32_16x16x32_bf16(
                false, a1.v, false, b1.v, (short)0, acc1, false, false);
        }
#pragma unroll
        for (int r = 0; r < 8; ++r) {
            const int row = m0 + (hi << 3) + r;
            const int col = n0 + lo;
            Dst[row * ldd + col] = f2b(fmaxf(acc0[r] + acc1[r] + bias[col], 0.f));
        }
    }
}

// ---------------------------------------------------------------------------
// Kernel 0: f32 -> bf16 conversion (weights + embedding tables), done once.
// ---------------------------------------------------------------------------
__global__ __launch_bounds__(256)
void cvt_bf16_kernel(const float* __restrict__ src, unsigned short* __restrict__ dst, int n) {
    int i = blockIdx.x * blockDim.x + threadIdx.x;
    const int stride = gridDim.x * blockDim.x;
    for (; i < n; i += stride) dst[i] = f2b(src[i]);
}

// ---------------------------------------------------------------------------
// Kernel 1: item_final for all B + B*LS slots. SPB slots per block;
// all three item-side weight matrices stay LDS-resident (staged once).
// ---------------------------------------------------------------------------
__global__ __launch_bounds__(128)
void item_final_kernel(const float* __restrict__ user_w,
                       const unsigned short* __restrict__ itemw_bf,
                       const unsigned short* __restrict__ ratew_bf,
                       const unsigned short* __restrict__ wln1_bf,
                       const unsigned short* __restrict__ watt1_bf,
                       const unsigned short* __restrict__ watt2_bf,
                       const float* __restrict__ ln1_b,
                       const float* __restrict__ ln2_w, const float* __restrict__ ln2_b,
                       const float* __restrict__ ln3_w, const float* __restrict__ ln3_b,
                       const float* __restrict__ att1_b, const float* __restrict__ att2_b,
                       const float* __restrict__ att3_w, const float* __restrict__ att3_b,
                       const int* __restrict__ nodes, const int* __restrict__ social_hist,
                       const int* __restrict__ item_hist, const int* __restrict__ rating_hist,
                       const int* __restrict__ item_mask,
                       float* __restrict__ hI_out) {
    const int tid = threadIdx.x;

    __shared__ unsigned short sW1[128 * 64];     // i_ln1  (16 KB)
    __shared__ unsigned short sA1[128 * 64];     // i_att1 (16 KB)
    __shared__ unsigned short sA2[64 * 64];      // i_att2 (8 KB)
    __shared__ unsigned short bufA[LII * 128];   // X0, later H|H2 (16 KB)
    __shared__ unsigned short bufX1[LII * 128];  // X1 | u (16 KB)
    __shared__ int   sIt[LII], sRt[LII], sMk[LII], sUid[SPB];
    __shared__ float sU[DD], sLog[LII], sAl[LII], sV1[DD], sV2[2 * DD];
    __shared__ float sRed;

    // Stage weights once per block (already bf16: plain copy).
    for (int i = tid; i < 128 * 64; i += 128) { sW1[i] = wln1_bf[i]; sA1[i] = watt1_bf[i]; }
    for (int i = tid; i < 64 * 64; i += 128) sA2[i] = watt2_bf[i];

    if (tid < SPB) {
        const int s = blockIdx.x * SPB + tid;
        int uid;
        if (s < BB) {
            uid = nodes[s];
        } else {
            const int t = s - BB;
            uid = social_hist[nodes[t / LSS] * LSS + (t % LSS)];
        }
        sUid[tid] = uid;
        __builtin_prefetch(&item_hist[(size_t)uid * LII], 0, 1);    // global_prefetch_b8
        __builtin_prefetch(&rating_hist[(size_t)uid * LII], 0, 1);
        __builtin_prefetch(&item_mask[(size_t)uid * LII], 0, 1);
    }
    __syncthreads();

    for (int s0 = 0; s0 < SPB; ++s0) {
        const int s = blockIdx.x * SPB + s0;
        const int uid = sUid[s0];

        if (tid < LII) {
            sIt[tid] = item_hist[uid * LII + tid];
            sRt[tid] = rating_hist[uid * LII + tid];
            sMk[tid] = item_mask[uid * LII + tid];
        }
        if (tid < DD) sU[tid] = user_w[uid * DD + tid];
        __syncthreads();

        // X0 = concat(item_w[hist], rating_w[hist]) [64 x 128] bf16 (copy only)
        for (int i = tid; i < LII * 128; i += 128) {
            const int l = i >> 7, c = i & 127;
            bufA[i] = (c < 64) ? itemw_bf[(size_t)sIt[l] * DD + c]
                               : ratew_bf[sRt[l] * DD + (c - 64)];
        }
        __syncthreads();

        // X1 = relu(X0 @ ln1 + b) -> left half of bufX1
        gemm_relu(bufA, 128, 64, 128, sW1, ln1_b, bufX1, 128, tid);
        __syncthreads();
        for (int i = tid; i < LII * 64; i += 128) {
            const int l = i >> 6, c = i & 63;
            bufX1[l * 128 + 64 + c] = f2b(sU[c]);
        }
        __syncthreads();

        // H = relu([X1|u] @ att1 + b) -> bufA[0..4096)
        gemm_relu(bufX1, 128, 64, 128, sA1, att1_b, bufA, 64, tid);
        __syncthreads();
        // H2 = relu(H @ att2 + b) -> bufA[4096..8192)
        gemm_relu(bufA, 64, 64, 64, sA2, att2_b, bufA + 64 * 64, 64, tid);
        __syncthreads();

        // logits + masked softmax over LI=64
        if (tid < LII) {
            float acc = att3_b[0];
            const unsigned short* h2 = bufA + 64 * 64 + tid * 64;
            for (int d = 0; d < 64; ++d) acc += b2f(h2[d]) * att3_w[d];
            sLog[tid] = sMk[tid] ? acc : NEGV;
        }
        __syncthreads();
        if (tid == 0) {
            float m = sLog[0];
            for (int l = 1; l < LII; ++l) m = fmaxf(m, sLog[l]);
            float sum = 0.f;
            for (int l = 0; l < LII; ++l) { float e = __expf(sLog[l] - m); sAl[l] = e; sum += e; }
            sRed = 1.f / sum;
        }
        __syncthreads();

        // hI = sum_l alpha_l * X1[l]
        if (tid < DD) {
            float acc = 0.f;
            for (int l = 0; l < LII; ++l) acc += sAl[l] * sRed * b2f(bufX1[l * 128 + tid]);
            sV1[tid] = acc;
        }
        __syncthreads();
        // hI = relu(hI @ ln2 + b); f = [u | hI]
        if (tid < DD) {
            float acc = ln2_b[tid];
            for (int k = 0; k < 64; ++k) acc += sV1[k] * ln2_w[k * 64 + tid];
            sV2[64 + tid] = fmaxf(acc, 0.f);
            sV2[tid] = sU[tid];
        }
        __syncthreads();
        // out = relu(f @ ln3 + b)
        if (tid < DD) {
            float acc = ln3_b[tid];
            for (int k = 0; k < 128; ++k) acc += sV2[k] * ln3_w[k * 64 + tid];
            hI_out[(size_t)s * DD + tid] = fmaxf(acc, 0.f);
        }
        __syncthreads();  // protect shared reuse across slots
    }
}

// ---------------------------------------------------------------------------
// Kernel 2: social aggregation + final MLPs. One block per batch node.
// ---------------------------------------------------------------------------
__global__ __launch_bounds__(128)
void social_kernel(const float* __restrict__ user_w,
                   const unsigned short* __restrict__ watt1_bf,
                   const unsigned short* __restrict__ watt2_bf,
                   const float* __restrict__ ln1_w, const float* __restrict__ ln1_b,
                   const float* __restrict__ ln2_w, const float* __restrict__ ln2_b,
                   const float* __restrict__ ln3_w, const float* __restrict__ ln3_b,
                   const float* __restrict__ att1_b, const float* __restrict__ att2_b,
                   const float* __restrict__ att3_w, const float* __restrict__ att3_b,
                   const int* __restrict__ nodes, const int* __restrict__ social_mask,
                   const float* __restrict__ hI, float* __restrict__ out) {
    const int b = blockIdx.x;
    const int tid = threadIdx.x;

    __shared__ unsigned short sW1[128 * 64];     // s_att1
    __shared__ unsigned short sW2[64 * 64];      // s_att2
    __shared__ unsigned short bufCat[LSS * 128]; // [hI_nb | u] bf16
    __shared__ unsigned short bufH[LSS * 64];
    __shared__ unsigned short bufH2[LSS * 64];
    __shared__ int   sMk[LSS];
    __shared__ float sU[DD], sLog[LSS], sAl[LSS], sV1[DD], sV2[2 * DD];
    __shared__ float sRed;

    const int uid = nodes[b];
    if (tid < DD) sU[tid] = user_w[uid * DD + tid];
    if (tid < LSS) sMk[tid] = social_mask[uid * LSS + tid];
    for (int i = tid; i < 128 * 64; i += 128) sW1[i] = watt1_bf[i];
    for (int i = tid; i < 64 * 64; i += 128) sW2[i] = watt2_bf[i];
    __syncthreads();

    for (int i = tid; i < LSS * 128; i += 128) {
        const int l = i >> 7, c = i & 127;
        bufCat[i] = (c < 64) ? f2b(hI[(size_t)(BB + b * LSS + l) * DD + c]) : f2b(sU[c - 64]);
    }
    __syncthreads();

    gemm_relu(bufCat, 128, 32, 128, sW1, att1_b, bufH, 64, tid);
    __syncthreads();
    gemm_relu(bufH, 64, 32, 64, sW2, att2_b, bufH2, 64, tid);
    __syncthreads();

    if (tid < LSS) {
        float acc = att3_b[0];
        for (int d = 0; d < 64; ++d) acc += b2f(bufH2[tid * 64 + d]) * att3_w[d];
        sLog[tid] = sMk[tid] ? acc : NEGV;
    }
    __syncthreads();
    if (tid == 0) {
        float m = sLog[0];
        for (int l = 1; l < LSS; ++l) m = fmaxf(m, sLog[l]);
        float sum = 0.f;
        for (int l = 0; l < LSS; ++l) { float e = __expf(sLog[l] - m); sAl[l] = e; sum += e; }
        sRed = 1.f / sum;
    }
    __syncthreads();

    if (tid < DD) {  // hS = sum_l beta_l * hI_nb[l]
        float acc = 0.f;
        for (int l = 0; l < LSS; ++l) acc += sAl[l] * sRed * b2f(bufCat[l * 128 + tid]);
        sV1[tid] = acc;
    }
    __syncthreads();
    if (tid < DD) {  // hS = relu(hS @ s_ln1 + b); f = [hI_self | hS]
        float acc = ln1_b[tid];
        for (int k = 0; k < 64; ++k) acc += sV1[k] * ln1_w[k * 64 + tid];
        sV2[64 + tid] = fmaxf(acc, 0.f);
        sV2[tid] = hI[(size_t)b * DD + tid];
    }
    __syncthreads();
    if (tid < DD) {  // f = relu(f @ s_ln2 + b)
        float acc = ln2_b[tid];
        for (int k = 0; k < 128; ++k) acc += sV2[k] * ln2_w[k * 64 + tid];
        sV1[tid] = fmaxf(acc, 0.f);
    }
    __syncthreads();
    if (tid < DD) {  // out = relu(f @ s_ln3 + b)
        float acc = ln3_b[tid];
        for (int k = 0; k < 64; ++k) acc += sV1[k] * ln3_w[k * 64 + tid];
        out[(size_t)b * DD + tid] = fmaxf(acc, 0.f);
    }
}

extern "C" void kernel_launch(void* const* d_in, const int* in_sizes, int n_in,
                              void* d_out, int out_size, void* d_ws, size_t ws_size,
                              hipStream_t stream) {
    const float* user_w   = (const float*)d_in[0];
    const float* item_w   = (const float*)d_in[1];
    const float* rating_w = (const float*)d_in[2];
    const float* i_ln1_w  = (const float*)d_in[3];  const float* i_ln1_b  = (const float*)d_in[4];
    const float* i_ln2_w  = (const float*)d_in[5];  const float* i_ln2_b  = (const float*)d_in[6];
    const float* i_ln3_w  = (const float*)d_in[7];  const float* i_ln3_b  = (const float*)d_in[8];
    const float* i_att1_w = (const float*)d_in[9];  const float* i_att1_b = (const float*)d_in[10];
    const float* i_att2_w = (const float*)d_in[11]; const float* i_att2_b = (const float*)d_in[12];
    const float* i_att3_w = (const float*)d_in[13]; const float* i_att3_b = (const float*)d_in[14];
    const float* s_ln1_w  = (const float*)d_in[15]; const float* s_ln1_b  = (const float*)d_in[16];
    const float* s_ln2_w  = (const float*)d_in[17]; const float* s_ln2_b  = (const float*)d_in[18];
    const float* s_ln3_w  = (const float*)d_in[19]; const float* s_ln3_b  = (const float*)d_in[20];
    const float* s_att1_w = (const float*)d_in[21]; const float* s_att1_b = (const float*)d_in[22];
    const float* s_att2_w = (const float*)d_in[23]; const float* s_att2_b = (const float*)d_in[24];
    const float* s_att3_w = (const float*)d_in[25]; const float* s_att3_b = (const float*)d_in[26];
    const int* nodes       = (const int*)d_in[27];
    const int* social_hist = (const int*)d_in[28];
    const int* social_mask = (const int*)d_in[29];
    const int* item_hist   = (const int*)d_in[30];
    const int* rating_hist = (const int*)d_in[31];
    const int* item_mask   = (const int*)d_in[32];

    const int nslots = BB * (1 + LSS);  // 16896

    // Workspace layout:
    //  [0, nslots*64 f32)                       hI results (4.33 MB)
    //  then bf16 (ushort) region:
    //    +0      i_ln1   (8192)
    //    +8192   i_att1  (8192)
    //    +16384  i_att2  (4096)
    //    +20480  s_att1  (8192)
    //    +28672  s_att2  (4096)
    //    +32768  rating_w (320, padded to 1024)
    //    +33792  item_w  (50000*64)
    float* hI = (float*)d_ws;
    unsigned short* bfbase =
        (unsigned short*)((char*)d_ws + (size_t)nslots * DD * sizeof(float));
    unsigned short* wln1_bf  = bfbase + 0;
    unsigned short* watt1_bf = bfbase + 8192;
    unsigned short* watt2_bf = bfbase + 16384;
    unsigned short* satt1_bf = bfbase + 20480;
    unsigned short* satt2_bf = bfbase + 28672;
    unsigned short* ratew_bf = bfbase + 32768;
    unsigned short* itemw_bf = bfbase + 33792;

    cvt_bf16_kernel<<<32, 256, 0, stream>>>(i_ln1_w, wln1_bf, 128 * 64);
    cvt_bf16_kernel<<<32, 256, 0, stream>>>(i_att1_w, watt1_bf, 128 * 64);
    cvt_bf16_kernel<<<16, 256, 0, stream>>>(i_att2_w, watt2_bf, 64 * 64);
    cvt_bf16_kernel<<<32, 256, 0, stream>>>(s_att1_w, satt1_bf, 128 * 64);
    cvt_bf16_kernel<<<16, 256, 0, stream>>>(s_att2_w, satt2_bf, 64 * 64);
    cvt_bf16_kernel<<<2, 256, 0, stream>>>(rating_w, ratew_bf, NRR * DD);
    cvt_bf16_kernel<<<2048, 256, 0, stream>>>(item_w, itemw_bf, NII * DD);

    item_final_kernel<<<nslots / SPB, 128, 0, stream>>>(
        user_w, itemw_bf, ratew_bf, wln1_bf, watt1_bf, watt2_bf,
        i_ln1_b, i_ln2_w, i_ln2_b, i_ln3_w, i_ln3_b,
        i_att1_b, i_att2_b, i_att3_w, i_att3_b,
        nodes, social_hist, item_hist, rating_hist, item_mask, hI);

    social_kernel<<<BB, 128, 0, stream>>>(
        user_w, satt1_bf, satt2_bf,
        s_ln1_w, s_ln1_b, s_ln2_w, s_ln2_b, s_ln3_w, s_ln3_b,
        s_att1_b, s_att2_b, s_att3_w, s_att3_b,
        nodes, social_mask, hI, (float*)d_out);
}
